// STFT_90159953478223
// MI455X (gfx1250) — matrix operand order repaired and checked by reading, not compiled, not run on hardware
//
#include <hip/hip_runtime.h>

// ---------------------------------------------------------------------------
// STFT as GEMM:  C[402, 60001] = basis[402,400] x frames[400,60001]
// frames[k,t] = sig_pad[t*HOP + k],  out[m*T + t] = C[m][t]
// bf16x3 split-precision GEMM on V_WMMA_F32_16X16X32_BF16 (f32 accumulate).
// ---------------------------------------------------------------------------

#define HOP    160
#define NFFT   400
#define CUT    201
#define MROWS  402          // 2*CUT rows of C
#define MP     416          // M padded to 16
#define KP     416          // K padded to 32
#define PAD    200          // CUT-1 zero padding each side of signal

typedef __attribute__((ext_vector_type(16))) __bf16 v16bf;
typedef __attribute__((ext_vector_type(8)))  float  v8f;

// ---- manual RNE float<->bf16 (avoid dependence on cast lowering) ----------
__device__ __forceinline__ __bf16 f2bf(float x) {
    unsigned u = __float_as_uint(x);
    unsigned r = (u + 0x7FFFu + ((u >> 16) & 1u)) >> 16;
    unsigned short s = (unsigned short)r;
    return *reinterpret_cast<__bf16*>(&s);
}
__device__ __forceinline__ float bf2f(__bf16 b) {
    unsigned short s = *reinterpret_cast<unsigned short*>(&b);
    unsigned u = ((unsigned)s) << 16;
    return __uint_as_float(u);
}

// ---- prep: pad signal with zeros, split into bf16 hi/lo planes ------------
__global__ void stft_split_signal(const float* __restrict__ sig,
                                  __bf16* __restrict__ hi, __bf16* __restrict__ lo,
                                  int t_sig, int total) {
    int i = blockIdx.x * blockDim.x + threadIdx.x;
    if (i >= total) return;
    float v = 0.0f;
    int j = i - PAD;
    if (j >= 0 && j < t_sig) v = sig[j];
    __bf16 h = f2bf(v);
    hi[i] = h;
    lo[i] = f2bf(v - bf2f(h));
}

// ---- prep: pad basis 402x400 -> 416x416 (zeros), split hi/lo --------------
__global__ void stft_split_basis(const float* __restrict__ basis,
                                 __bf16* __restrict__ hi, __bf16* __restrict__ lo) {
    int i = blockIdx.x * blockDim.x + threadIdx.x;
    if (i >= MP * KP) return;
    int m = i / KP, k = i % KP;
    float v = (m < MROWS && k < NFFT) ? basis[m * NFFT + k] : 0.0f;
    __bf16 h = f2bf(v);
    hi[i] = h;
    lo[i] = f2bf(v - bf2f(h));
}

__device__ __forceinline__ v8f wmma_bf16(v16bf a, v16bf b, v8f c) {
    // (neg_a, A, neg_b, B, c_mod, C, reuse_a, reuse_b)
    return __builtin_amdgcn_wmma_f32_16x16x32_bf16(false, a, false, b, (short)0, c,
                                                   false, false);
}

__device__ __forceinline__ void load_v16(v16bf& d, const __bf16* p0, const __bf16* p1) {
    // two 16-byte (8 x bf16) chunks -> one v16bf fragment
    *reinterpret_cast<uint4*>(&d)     = *reinterpret_cast<const uint4*>(p0);
    *(reinterpret_cast<uint4*>(&d)+1) = *reinterpret_cast<const uint4*>(p1);
}

// ---- main GEMM: grid = (ceil(T/128), MP/32), block = 256 (8 waves) --------
__global__ __launch_bounds__(256)
void stft_wmma_gemm(const __bf16* __restrict__ sig_hi, const __bf16* __restrict__ sig_lo,
                    const __bf16* __restrict__ bas_hi, const __bf16* __restrict__ bas_lo,
                    float* __restrict__ out, int T) {
    const int lane  = threadIdx.x & 31;
    const int wave  = threadIdx.x >> 5;
    const int mw    = wave & 1;            // 2 M-subtiles per WG
    const int nw    = wave >> 1;           // 4 N-subtiles per WG
    const int m0    = blockIdx.y * 32 + mw * 16;
    const int t0    = blockIdx.x * 128 + nw * 32;   // wave owns cols [t0, t0+32)

    const int lhalf = lane >> 4;           // 0|1
    const int lmod  = lane & 15;

    // A fragment (16x32 bf16): lane m = m0+lmod holds K = 16*lhalf .. +15 (contiguous)
    const __bf16* ah_p = bas_hi + (size_t)(m0 + lmod) * KP + 16 * lhalf;
    const __bf16* al_p = bas_lo + (size_t)(m0 + lmod) * KP + 16 * lhalf;

    // B fragment (32x16 bf16): lane col t = t?+lmod holds K = 16*lhalf .. +15 (contiguous!)
    const int ta = min(t0      + lmod, T - 1);     // clamp OOB columns to a valid address
    const int tb = min(t0 + 16 + lmod, T - 1);
    const __bf16* bha = sig_hi + (size_t)ta * HOP + 16 * lhalf;
    const __bf16* bla = sig_lo + (size_t)ta * HOP + 16 * lhalf;
    const __bf16* bhb = sig_hi + (size_t)tb * HOP + 16 * lhalf;
    const __bf16* blb = sig_lo + (size_t)tb * HOP + 16 * lhalf;

    v8f acc0 = {};   // cols [t0, t0+16)
    v8f acc1 = {};   // cols [t0+16, t0+32)

    for (int kc = 0; kc < KP; kc += 32) {
        v16bf ah, al, bh0, bl0, bh1, bl1;
        load_v16(ah,  ah_p + kc, ah_p + kc + 8);
        load_v16(al,  al_p + kc, al_p + kc + 8);
        load_v16(bh0, bha + kc,  bha + kc + 8);
        load_v16(bl0, bla + kc,  bla + kc + 8);
        load_v16(bh1, bhb + kc,  bhb + kc + 8);
        load_v16(bl1, blb + kc,  blb + kc + 8);

        // bf16x3: hi*hi + hi*lo + lo*hi (lo*lo < 2^-16, dropped); interleave tiles
        acc0 = wmma_bf16(ah, bh0, acc0);
        acc1 = wmma_bf16(ah, bh1, acc1);
        acc0 = wmma_bf16(ah, bl0, acc0);
        acc1 = wmma_bf16(ah, bl1, acc1);
        acc0 = wmma_bf16(al, bh0, acc0);
        acc1 = wmma_bf16(al, bh1, acc1);
    }

    // C/D layout: VGPR r -> M = r (lanes 0-15) | r+8 (lanes 16-31); N = lane&15
    const int n0 = t0 + lmod;
    const int n1 = t0 + 16 + lmod;
    const int mb = m0 + 8 * lhalf;
#pragma unroll
    for (int r = 0; r < 8; ++r) {
        int m = mb + r;
        if (m < MROWS) {
            if (n0 < T) out[(size_t)m * T + n0] = acc0[r];
            if (n1 < T) out[(size_t)m * T + n1] = acc1[r];
        }
    }
}

extern "C" void kernel_launch(void* const* d_in, const int* in_sizes, int n_in,
                              void* d_out, int out_size, void* d_ws, size_t ws_size,
                              hipStream_t stream) {
    const float* signal = (const float*)d_in[0];   // T_SIG f32
    const float* basis  = (const float*)d_in[1];   // 402*400 f32
    float* out = (float*)d_out;                    // 2*201*T f32 (== C[402][T] flat)

    const int t_sig       = in_sizes[0];
    const int sig_pad_len = t_sig + 2 * PAD + 16;  // +16 slack for K padded to 416
    const int T           = t_sig / HOP + 1;       // 60001

    // workspace layout (all 32B aligned)
    char* ws = (char*)d_ws;
    size_t off = 0;
    auto take = [&](size_t bytes) { char* p = ws + off; off = (off + bytes + 31) & ~(size_t)31; return p; };
    __bf16* sig_hi = (__bf16*)take((size_t)sig_pad_len * 2);
    __bf16* sig_lo = (__bf16*)take((size_t)sig_pad_len * 2);
    __bf16* bas_hi = (__bf16*)take((size_t)MP * KP * 2);
    __bf16* bas_lo = (__bf16*)take((size_t)MP * KP * 2);
    (void)ws_size; (void)n_in; (void)out_size;     // ~39.2 MB total required

    stft_split_signal<<<(sig_pad_len + 255) / 256, 256, 0, stream>>>(
        signal, sig_hi, sig_lo, t_sig, sig_pad_len);
    stft_split_basis<<<(MP * KP + 255) / 256, 256, 0, stream>>>(basis, bas_hi, bas_lo);

    dim3 grid((T + 127) / 128, MP / 32);           // (469, 13)
    stft_wmma_gemm<<<grid, 256, 0, stream>>>(sig_hi, sig_lo, bas_hi, bas_lo, out, T);
}
